// Thalamus_32366873543104
// MI455X (gfx1250) — matrix operand order repaired and checked by reading, not compiled, not run on hardware
//
#include <hip/hip_runtime.h>
#include <hip/hip_bf16.h>
#include <math.h>

// ---------------------------------------------------------------------------
// Mamba block + MoE top-2 router for MI455X (gfx1250, wave32, WMMA).
// GEMM operands pre-converted to bf16 (weights pre-transposed to N-major) so
// tile staging is a pure byte copy -> GLOBAL_LOAD_ASYNC_TO_LDS_B128 with
// double-buffered LDS and s_wait_asynccnt pipelining. WMMA f32 accumulate.
// ---------------------------------------------------------------------------

#define D_MODEL 768
#define D_INNER 1536
#define D_STATE 16
#define D_CONV  4
#define DT_RANK 48
#define N_EXP   8
#define BATCH   2
#define SEQLEN  2048
#define NTOK    (BATCH * SEQLEN)   // 4096

typedef __bf16 bf16_t;
typedef __attribute__((ext_vector_type(16))) __bf16 v16bf;
typedef __attribute__((ext_vector_type(8)))  __bf16 v8bf;
typedef __attribute__((ext_vector_type(4)))  __bf16 v4bf;
typedef __attribute__((ext_vector_type(8)))  float  v8f;

// LDS pitch in bf16 elements: 80-byte rows -> 16B aligned chunks, and the
// 20-dword pitch puts 16 lanes on disjoint 4-bank groups (conflict-free b128).
#define AS_LD 40   // As: 64 rows (M) x 32 (K)
#define BT_LD 40   // Bs: 64 rows (N) x 32 (K)  (B pre-transposed in global)

// ---- CDNA5 async copy helpers (ISA 10., ASYNCcnt) -------------------------
__device__ __forceinline__ void async_b128_to_lds(void* lds_ptr, const void* gptr) {
    unsigned lds_off = (unsigned)(size_t)lds_ptr;   // LDS aperture: addr[31:0]
    asm volatile("global_load_async_to_lds_b128 %0, %1, off"
                 :: "v"(lds_off), "v"(gptr) : "memory");
}
__device__ __forceinline__ void wait_async_le2() {
    asm volatile("s_wait_asynccnt 0x2" ::: "memory");
}
__device__ __forceinline__ void wait_async_0() {
    asm volatile("s_wait_asynccnt 0x0" ::: "memory");
}

// ---- WMMA fragment loads from LDS (ISA 7.12.2 layouts) --------------------
// A 16x32: lanes 0-15 M=lane, K {0..7, 16..23}; lanes 16-31 M=lane-16,
//          K {8..15, 24..31}  -> two b128 chunks 32 bytes apart.
__device__ __forceinline__ v16bf frag_a(const bf16_t* p) {
    v8bf a0 = *(const v8bf*)(p);
    v8bf a1 = *(const v8bf*)(p + 16);
    return __builtin_shufflevector(a0, a1, 0,1,2,3,4,5,6,7,8,9,10,11,12,13,14,15);
}
// B 32x16: lanes 0-15 N=lane K {0..7, 16..23}; lanes 16-31 N=lane-16
//          K {8..15, 24..31}  -> two b128 chunks 32 bytes apart.
__device__ __forceinline__ v16bf frag_b(const bf16_t* p) {
    v8bf b0 = *(const v8bf*)(p);
    v8bf b1 = *(const v8bf*)(p + 16);
    return __builtin_shufflevector(b0, b1, 0,1,2,3,4,5,6,7,8,9,10,11,12,13,14,15);
}

__device__ __forceinline__ float softplusf(float v) {
    return (v > 20.0f) ? v : log1pf(expf(v));
}

// ---------------------------------------------------------------------------
// Aligned GEMM: C[M,N] = act(A * Bt^T + bias). A: bf16 M x K (lda). Bt: bf16
// N x K (ldbt, pre-transposed). Requires M%64==0, N%64==0, K%32==0.
// 256 thr = 8 waves 4(M)x2(N); block tile 64x64; wave tile 16x32 = 2 wmma
// sharing the A fragment. Double-buffered LDS, async tile DMA.
// ---------------------------------------------------------------------------
__global__ __launch_bounds__(256)
void gemm_wmma_async(const bf16_t* __restrict__ A, const bf16_t* __restrict__ Bt,
                     const float* __restrict__ bias, float* __restrict__ C,
                     int M, int N, int K, int lda, int ldbt, int ldc, int act)
{
    __shared__ bf16_t As[2][64 * AS_LD];
    __shared__ bf16_t Bs[2][64 * BT_LD];

    const int tid  = threadIdx.x;
    const int lane = tid & 31;
    const int wave = tid >> 5;
    const int wm   = wave & 3;
    const int wn   = wave >> 2;
    const int blockM = blockIdx.y * 64;
    const int blockN = blockIdx.x * 64;

    const bool lo = lane < 16;
    const int  hl = lo ? lane : lane - 16;

    // per-thread slice of the 64x32 tile: 8 bf16 = 16 bytes
    const int e  = tid * 8;
    const int tr = e >> 5;        // tile row (M for A, N for Bt)
    const int tc = e & 31;        // tile col (K)

    v8f acc0 = {};
    v8f acc1 = {};

    // software pipeline: stage k-tile `k0` into buffer `buf`
    auto stage = [&](int buf, int k0) {
        async_b128_to_lds(&As[buf][tr * AS_LD + tc],
                          A  + (size_t)(blockM + tr) * lda  + (k0 + tc));
        async_b128_to_lds(&Bs[buf][tr * BT_LD + tc],
                          Bt + (size_t)(blockN + tr) * ldbt + (k0 + tc));
    };

    stage(0, 0);
    int it = 0;
    for (int k0 = 0; k0 < K; k0 += 32, ++it) {
        const int cur = it & 1;
        const bool has_next = (k0 + 32) < K;
        if (has_next) {
            stage(cur ^ 1, k0 + 32);   // next tile DMA overlaps this tile's math
            wait_async_le2();          // in-order: <=2 outstanding => cur landed
        } else {
            wait_async_0();
        }
        __syncthreads();

        v16bf afrag = frag_a(&As[cur][(wm * 16 + hl) * AS_LD + (lo ? 0 : 8)]);
        const int kb2 = lo ? 0 : 8;
        v16bf b0 = frag_b(&Bs[cur][(wn * 32 +      hl) * BT_LD + kb2]);
        v16bf b1 = frag_b(&Bs[cur][(wn * 32 + 16 + hl) * BT_LD + kb2]);
        acc0 = __builtin_amdgcn_wmma_f32_16x16x32_bf16(
                   false, afrag, false, b0, (short)0, acc0, false, false);
        acc1 = __builtin_amdgcn_wmma_f32_16x16x32_bf16(
                   false, afrag, false, b1, (short)0, acc1, false, false);
        __syncthreads();
    }

    #pragma unroll
    for (int s = 0; s < 2; ++s) {
        const v8f& acc = s ? acc1 : acc0;
        const int colN = blockN + wn * 32 + s * 16 + hl;
        const float bv = bias ? bias[colN] : 0.0f;
        #pragma unroll
        for (int r = 0; r < 8; ++r) {
            int rowM = blockM + wm * 16 + r + (lo ? 0 : 8);
            float v = acc[r] + bv;
            if (act == 1) v = softplusf(v);
            C[(size_t)rowM * ldc + colN] = v;
        }
    }
}

// ---------------------------------------------------------------------------
// Guarded GEMM for the small tail cases (N=80 or K=48): synchronous staging,
// v4bf row copies with zero-fill, optional secondary bf16 output (Cbf).
// Requires M%64==0 and K,N multiples of 4.
// ---------------------------------------------------------------------------
__global__ __launch_bounds__(256)
void gemm_wmma_guard(const bf16_t* __restrict__ A, const bf16_t* __restrict__ Bt,
                     const float* __restrict__ bias, float* __restrict__ C,
                     bf16_t* __restrict__ Cbf,
                     int M, int N, int K, int lda, int ldbt, int ldc, int act)
{
    __shared__ bf16_t As[64 * AS_LD];
    __shared__ bf16_t Bs[64 * BT_LD];

    const int tid  = threadIdx.x;
    const int lane = tid & 31;
    const int wave = tid >> 5;
    const int wm   = wave & 3;
    const int wn   = wave >> 2;
    const int blockM = blockIdx.y * 64;
    const int blockN = blockIdx.x * 64;

    const bool lo = lane < 16;
    const int  hl = lo ? lane : lane - 16;

    v8f acc0 = {};
    v8f acc1 = {};

    const int Kt = (K + 31) & ~31;
    for (int k0 = 0; k0 < Kt; k0 += 32) {
        #pragma unroll
        for (int i = 0; i < 2; ++i) {
            int e = tid * 4 + i * 1024;      // 0..2047
            int r = e >> 5;                  // tile row
            int c = e & 31;                  // K col
            int gk = k0 + c;
            v4bf va = {};
            if (gk + 4 <= K)
                va = *(const v4bf*)(A + (size_t)(blockM + r) * lda + gk);
            *(v4bf*)&As[r * AS_LD + c] = va;
            v4bf vb = {};
            if ((blockN + r) < N && gk + 4 <= K)
                vb = *(const v4bf*)(Bt + (size_t)(blockN + r) * ldbt + gk);
            *(v4bf*)&Bs[r * BT_LD + c] = vb;
        }
        __syncthreads();

        v16bf afrag = frag_a(&As[(wm * 16 + hl) * AS_LD + (lo ? 0 : 8)]);
        const int kb2 = lo ? 0 : 8;
        v16bf b0 = frag_b(&Bs[(wn * 32 +      hl) * BT_LD + kb2]);
        v16bf b1 = frag_b(&Bs[(wn * 32 + 16 + hl) * BT_LD + kb2]);
        acc0 = __builtin_amdgcn_wmma_f32_16x16x32_bf16(
                   false, afrag, false, b0, (short)0, acc0, false, false);
        acc1 = __builtin_amdgcn_wmma_f32_16x16x32_bf16(
                   false, afrag, false, b1, (short)0, acc1, false, false);
        __syncthreads();
    }

    #pragma unroll
    for (int s = 0; s < 2; ++s) {
        const v8f& acc = s ? acc1 : acc0;
        const int colN = blockN + wn * 32 + s * 16 + hl;
        if (colN < N) {
            const float bv = bias ? bias[colN] : 0.0f;
            #pragma unroll
            for (int r = 0; r < 8; ++r) {
                int rowM = blockM + wm * 16 + r + (lo ? 0 : 8);
                float v = acc[r] + bv;
                if (act == 1) v = softplusf(v);
                C[(size_t)rowM * ldc + colN] = v;
                if (Cbf) Cbf[(size_t)rowM * ldc + colN] = (bf16_t)v;
            }
        }
    }
}

// ---------------------------------------------------------------------------
// Pre-passes: f32 -> bf16 convert; f32 KxN -> bf16 NxK tiled transpose.
// ---------------------------------------------------------------------------
__global__ __launch_bounds__(256)
void convert_bf16_kernel(const float* __restrict__ src, bf16_t* __restrict__ dst, int n)
{
    int i = blockIdx.x * blockDim.x + threadIdx.x;
    if (i < n) dst[i] = (bf16_t)src[i];
}

__global__ __launch_bounds__(256)
void transpose_bf16_kernel(const float* __restrict__ W, bf16_t* __restrict__ Wt,
                           int K, int N)   // W: KxN, Wt: NxK
{
    __shared__ float tile[32][33];
    int kb = blockIdx.x * 32, nb = blockIdx.y * 32;
    int tx = threadIdx.x & 31, ty = threadIdx.x >> 5;   // ty: 0..7
    #pragma unroll
    for (int j = 0; j < 32; j += 8) {
        int k = kb + ty + j, n = nb + tx;
        tile[ty + j][tx] = (k < K && n < N) ? W[(size_t)k * N + n] : 0.0f;
    }
    __syncthreads();
    #pragma unroll
    for (int j = 0; j < 32; j += 8) {
        int n = nb + ty + j, k = kb + tx;
        if (n < N && k < K) Wt[(size_t)n * K + k] = (bf16_t)tile[tx][ty + j];
    }
}

// ---------------------------------------------------------------------------
// Causal depthwise conv1d (K=4) + bias + SiLU; writes f32 (scan) + bf16 (GEMM).
// ---------------------------------------------------------------------------
__global__ __launch_bounds__(256)
void conv_silu_kernel(const float* __restrict__ xz,
                      const float* __restrict__ conv_w,
                      const float* __restrict__ conv_b,
                      float* __restrict__ x_act, bf16_t* __restrict__ x_act_bf)
{
    int idx = blockIdx.x * blockDim.x + threadIdx.x;
    if (idx >= NTOK * D_INNER) return;
    int d = idx % D_INNER;
    int t = idx / D_INNER;
    int l = t % SEQLEN;
    int b = t / SEQLEN;

    float s = conv_b[d];
    #pragma unroll
    for (int j = 0; j < D_CONV; ++j) {
        int ll = l - (D_CONV - 1) + j;
        if (ll >= 0)
            s += conv_w[d * D_CONV + j] *
                 xz[((size_t)(b * SEQLEN + ll)) * (2 * D_INNER) + d];
    }
    float v = s / (1.0f + expf(-s));    // silu
    x_act[idx]    = v;
    x_act_bf[idx] = (bf16_t)v;
}

// ---------------------------------------------------------------------------
// Selective scan: one thread per (batch, channel), 16 states in registers.
// Emits gated y as bf16 (only consumed by the output GEMM).
// ---------------------------------------------------------------------------
__global__ __launch_bounds__(256)
void scan_kernel(const float* __restrict__ delta,
                 const float* __restrict__ x_act,
                 const float* __restrict__ dbl,      // (T, 80): dt|B|C
                 const float* __restrict__ xz,       // z gate
                 const float* __restrict__ A_log,
                 const float* __restrict__ Dvec,
                 bf16_t* __restrict__ y_bf)
{
    int id = blockIdx.x * blockDim.x + threadIdx.x;
    if (id >= BATCH * D_INNER) return;
    int d = id % D_INNER;
    int b = id / D_INNER;

    float Arow[D_STATE];
    #pragma unroll
    for (int n = 0; n < D_STATE; ++n)
        Arow[n] = -expf(A_log[d * D_STATE + n]);
    const float Dd = Dvec[d];

    float h[D_STATE];
    #pragma unroll
    for (int n = 0; n < D_STATE; ++n) h[n] = 0.0f;

    for (int t = 0; t < SEQLEN; ++t) {
        size_t tt = (size_t)b * SEQLEN + t;
        float dl = delta[tt * D_INNER + d];
        float xa = x_act[tt * D_INNER + d];
        float dx = dl * xa;
        const float* row = dbl + tt * (DT_RANK + 2 * D_STATE);
        float accv = 0.0f;
        #pragma unroll
        for (int n = 0; n < D_STATE; ++n) {
            float dA = expf(dl * Arow[n]);
            h[n] = dA * h[n] + dx * row[DT_RANK + n];          // B
            accv += h[n] * row[DT_RANK + D_STATE + n];         // C
        }
        accv += Dd * xa;
        float zz = xz[tt * (2 * D_INNER) + D_INNER + d];
        float g = zz / (1.0f + expf(-zz));                     // silu(z)
        y_bf[tt * D_INNER + d] = (bf16_t)(accv * g);
    }
}

// ---------------------------------------------------------------------------
// Router: logits = ctx @ W_r + b_r, softmax, top-2, renormalize.
// ---------------------------------------------------------------------------
__global__ __launch_bounds__(256)
void router_kernel(const float* __restrict__ ctx,
                   const float* __restrict__ W_r,
                   const float* __restrict__ b_r,
                   float* __restrict__ out_w,
                   int* __restrict__ out_idx)
{
    int t = blockIdx.x * blockDim.x + threadIdx.x;
    if (t >= NTOK) return;

    float logit[N_EXP];
    #pragma unroll
    for (int e = 0; e < N_EXP; ++e) logit[e] = b_r[e];
    const float* row = ctx + (size_t)t * D_MODEL;
    for (int k = 0; k < D_MODEL; ++k) {
        float xv = row[k];
        #pragma unroll
        for (int e = 0; e < N_EXP; ++e)
            logit[e] += xv * W_r[k * N_EXP + e];
    }
    float mx = logit[0];
    #pragma unroll
    for (int e = 1; e < N_EXP; ++e) mx = fmaxf(mx, logit[e]);
    float p[N_EXP], sum = 0.0f;
    #pragma unroll
    for (int e = 0; e < N_EXP; ++e) { p[e] = expf(logit[e] - mx); sum += p[e]; }

    int i0 = 0;
    #pragma unroll
    for (int e = 1; e < N_EXP; ++e) if (p[e] > p[i0]) i0 = e;
    int i1 = (i0 == 0) ? 1 : 0;
    #pragma unroll
    for (int e = 0; e < N_EXP; ++e) if (e != i0 && p[e] > p[i1]) i1 = e;

    float w0 = p[i0] / sum, w1 = p[i1] / sum;
    float tw = w0 + w1;
    out_w[t * 2 + 0] = w0 / tw;
    out_w[t * 2 + 1] = w1 / tw;
    out_idx[t * 2 + 0] = i0;
    out_idx[t * 2 + 1] = i1;
}

// ---------------------------------------------------------------------------
// Launch. Inputs: 0 x, 1 W_in, 2 conv_w, 3 conv_b, 4 W_x, 5 W_dt, 6 b_dt,
// 7 A_log, 8 D, 9 W_out, 10 W_r, 11 b_r.
// Output: routing_weights (8192 f32) then top_idx (8192 i32).
// ---------------------------------------------------------------------------
extern "C" void kernel_launch(void* const* d_in, const int* in_sizes, int n_in,
                              void* d_out, int out_size, void* d_ws, size_t ws_size,
                              hipStream_t stream)
{
    const float* x      = (const float*)d_in[0];
    const float* W_in   = (const float*)d_in[1];
    const float* conv_w = (const float*)d_in[2];
    const float* conv_b = (const float*)d_in[3];
    const float* W_x    = (const float*)d_in[4];
    const float* W_dt   = (const float*)d_in[5];
    const float* b_dt   = (const float*)d_in[6];
    const float* A_log  = (const float*)d_in[7];
    const float* Dvec   = (const float*)d_in[8];
    const float* W_out  = (const float*)d_in[9];
    const float* W_r    = (const float*)d_in[10];
    const float* b_r    = (const float*)d_in[11];

    const int DBL = DT_RANK + 2 * D_STATE;   // 80

    // ---- workspace carve-out (all offsets 256B aligned) ----
    char* ws = (char*)d_ws;
    size_t off = 0;
    auto carve = [&](size_t bytes) {
        void* p = ws + off;
        off += (bytes + 255) & ~(size_t)255;
        return p;
    };
    float*  xz       = (float*) carve((size_t)NTOK * 2 * D_INNER * 4);  // 48 MB
    float*  x_act    = (float*) carve((size_t)NTOK * D_INNER * 4);      // 24 MB
    float*  delta    = (float*) carve((size_t)NTOK * D_INNER * 4);      // 24 MB
    float*  dbl      = (float*) carve((size_t)NTOK * DBL * 4);
    bf16_t* dbl_bf   = (bf16_t*)carve((size_t)NTOK * DBL * 2);
    bf16_t* x_bf     = (bf16_t*)carve((size_t)NTOK * D_MODEL * 2);
    bf16_t* x_act_bf = (bf16_t*)carve((size_t)NTOK * D_INNER * 2);
    bf16_t* y_bf     = (bf16_t*)carve((size_t)NTOK * D_INNER * 2);
    bf16_t* Wt_in    = (bf16_t*)carve((size_t)2 * D_INNER * D_MODEL * 2);
    bf16_t* Wt_x     = (bf16_t*)carve((size_t)DBL * D_INNER * 2);
    bf16_t* Wt_dt    = (bf16_t*)carve((size_t)D_INNER * DT_RANK * 2);
    bf16_t* Wt_out   = (bf16_t*)carve((size_t)D_MODEL * D_INNER * 2);
    float*  ctx      = x_act;    // x_act dead after scan

    // ---- 0) operand conversion / weight transpose pre-passes ----
    convert_bf16_kernel<<<(NTOK * D_MODEL + 255) / 256, 256, 0, stream>>>(
        x, x_bf, NTOK * D_MODEL);
    transpose_bf16_kernel<<<dim3((D_MODEL + 31) / 32, (2 * D_INNER + 31) / 32), 256, 0, stream>>>(
        W_in, Wt_in, D_MODEL, 2 * D_INNER);
    transpose_bf16_kernel<<<dim3((D_INNER + 31) / 32, (DBL + 31) / 32), 256, 0, stream>>>(
        W_x, Wt_x, D_INNER, DBL);
    transpose_bf16_kernel<<<dim3((DT_RANK + 31) / 32, (D_INNER + 31) / 32), 256, 0, stream>>>(
        W_dt, Wt_dt, DT_RANK, D_INNER);
    transpose_bf16_kernel<<<dim3((D_INNER + 31) / 32, (D_MODEL + 31) / 32), 256, 0, stream>>>(
        W_out, Wt_out, D_INNER, D_MODEL);

    // ---- 1) xz = x @ W_in           (4096x768 @ 768x3072), async path ----
    gemm_wmma_async<<<dim3((2 * D_INNER) / 64, NTOK / 64), 256, 0, stream>>>(
        x_bf, Wt_in, nullptr, xz,
        NTOK, 2 * D_INNER, D_MODEL, D_MODEL, D_MODEL, 2 * D_INNER, 0);

    // ---- 2) x_act = silu(causal_conv(xi) + conv_b) ----
    conv_silu_kernel<<<(NTOK * D_INNER + 255) / 256, 256, 0, stream>>>(
        xz, conv_w, conv_b, x_act, x_act_bf);

    // ---- 3) dbl = x_act @ W_x       (N=80 tail), dual f32/bf16 output ----
    gemm_wmma_guard<<<dim3((DBL + 63) / 64, NTOK / 64), 256, 0, stream>>>(
        x_act_bf, Wt_x, nullptr, dbl, dbl_bf,
        NTOK, DBL, D_INNER, D_INNER, D_INNER, DBL, 0);

    // ---- 4) delta = softplus(dt @ W_dt + b_dt)   (K=48 tail) ----
    gemm_wmma_guard<<<dim3(D_INNER / 64, NTOK / 64), 256, 0, stream>>>(
        dbl_bf, Wt_dt, b_dt, delta, nullptr,
        NTOK, D_INNER, DT_RANK, DBL, DT_RANK, D_INNER, 1);

    // ---- 5) selective scan + D skip + silu(z) gate -> y (bf16) ----
    scan_kernel<<<(BATCH * D_INNER + 255) / 256, 256, 0, stream>>>(
        delta, x_act, dbl, xz, A_log, Dvec, y_bf);

    // ---- 6) ctx = y @ W_out          (4096x1536 @ 1536x768), async path ----
    gemm_wmma_async<<<dim3(D_MODEL / 64, NTOK / 64), 256, 0, stream>>>(
        y_bf, Wt_out, nullptr, ctx,
        NTOK, D_MODEL, D_INNER, D_INNER, D_INNER, D_MODEL, 0);

    // ---- 7) router ----
    float* out_w   = (float*)d_out;
    int*   out_idx = (int*)((float*)d_out + (size_t)NTOK * 2);
    router_kernel<<<(NTOK + 255) / 256, 256, 0, stream>>>(
        ctx, W_r, b_r, out_w, out_idx);
}
